// MultiHeadAttention_14766097563889
// MI455X (gfx1250) — compile-verified, matrix operand order checked by reading, not run on hardware
//
#include <hip/hip_runtime.h>

// ---------------------------------------------------------------------------
// MI455X (gfx1250) fused multi-head attention, bf16 WMMA + fp32 accumulation.
//   B=2, T=2048, C=1024, H=16, D=64   (wave32, v_wmma_f32_16x16x32_bf16)
// K0: RoPE cos/sin table (2048 x 32, fp32 pairs) -> L2-resident
// K1: QKV GEMM + fused RoPE/scale -> bf16 (mat,B,H,T,D)
// K2: flash attention; K tiles DMA'd by the Tensor Data Mover into a
//     double-buffered LDS slot (s_wait_tensorcnt pipelining), V tiles
//     SW-transposed. All WMMA fragment loads are 2x ds_load_b128.
// K3: output projection + bias -> fp32 (A tile via TDM)
// ---------------------------------------------------------------------------

typedef __bf16 bf16_t;
typedef __attribute__((ext_vector_type(16))) __bf16 v16bf;
typedef __attribute__((ext_vector_type(8)))  float  v8f;
typedef __attribute__((ext_vector_type(4))) unsigned int u32x4;
typedef __attribute__((ext_vector_type(8))) int i32x8;
typedef __attribute__((ext_vector_type(4))) int i32x4;
typedef unsigned int u32;
typedef unsigned long long u64;

#define LDSS 80   // padded LDS row stride (elems): 160B rows, 16B aligned

#if defined(__has_builtin)
#if __has_builtin(__builtin_amdgcn_tensor_load_to_lds)
#define HAVE_TDM 1
#endif
#endif

__device__ __forceinline__ v8f wmma_bf16(v16bf a, v16bf b, v8f c) {
  return __builtin_amdgcn_wmma_f32_16x16x32_bf16(false, a, false, b, (short)0, c,
                                                 false, false);
}

__device__ __forceinline__ u32 pack2(float lo, float hi) {
  union { bf16_t h[2]; u32 u; } w;
  w.h[0] = (bf16_t)lo; w.h[1] = (bf16_t)hi;
  return w.u;
}

// Fragment load: lane&15 selects the operand row (A rows, or B^T columns);
// per ISA 7.12.2 a lane's 16 bf16 are two contiguous 8-element runs at
// K = kh..kh+7 and K = 16+kh..16+kh+7  ->  two ds_load_b128.
__device__ __forceinline__ v16bf load_frag(const bf16_t* s, int stride) {
  const int lane = threadIdx.x & 31;
  const int m  = lane & 15;
  const int kh = (lane >> 4) << 3;
  union { uint4 u[2]; v16bf v; } f;
  f.u[0] = *(const uint4*)(s + m * stride + kh);
  f.u[1] = *(const uint4*)(s + m * stride + 16 + kh);
  return f.v;
}

// --- staging helpers (128 threads, 64x64 tiles) ----------------------------

__device__ __forceinline__ void stage_f32_rm(const float* __restrict__ g, int ldg,
                                             bf16_t* __restrict__ s) {
  const int t    = threadIdx.x;
  const int row  = t >> 1;
  const int colb = (t & 1) << 5;
  const float4* gp = (const float4*)(g + (size_t)row * ldg + colb);
  bf16_t* sp = s + row * LDSS + colb;
#pragma unroll
  for (int i = 0; i < 8; ++i) {
    float4 v = gp[i];
    *(u32*)(sp + 4 * i)     = pack2(v.x, v.y);
    *(u32*)(sp + 4 * i + 2) = pack2(v.z, v.w);
  }
}

__device__ __forceinline__ void stage_f32_T(const float* __restrict__ g, int ldg,
                                            bf16_t* __restrict__ sT) {
  const int t  = threadIdx.x;
  const int p  = t >> 2;               // K row pair (2p, 2p+1)
  const int cb = (t & 3) << 4;         // 16-wide column chunk
  const float4* g0 = (const float4*)(g + (size_t)(2 * p) * ldg + cb);
  const float4* g1 = (const float4*)(g + (size_t)(2 * p + 1) * ldg + cb);
#pragma unroll
  for (int q = 0; q < 4; ++q) {
    float4 a = g0[q], b = g1[q];
    *(u32*)(sT + (size_t)(cb + 4 * q + 0) * LDSS + 2 * p) = pack2(a.x, b.x);
    *(u32*)(sT + (size_t)(cb + 4 * q + 1) * LDSS + 2 * p) = pack2(a.y, b.y);
    *(u32*)(sT + (size_t)(cb + 4 * q + 2) * LDSS + 2 * p) = pack2(a.z, b.z);
    *(u32*)(sT + (size_t)(cb + 4 * q + 3) * LDSS + 2 * p) = pack2(a.w, b.w);
  }
}

__device__ __forceinline__ void stage_bf16_rm(const bf16_t* __restrict__ g, int ldg,
                                              bf16_t* __restrict__ s) {
  const int t    = threadIdx.x;
  const int row  = t >> 1;
  const int colb = (t & 1) << 5;
  const uint4* gp = (const uint4*)(g + (size_t)row * ldg + colb);
  uint4* sp = (uint4*)(s + row * LDSS + colb);
#pragma unroll
  for (int i = 0; i < 4; ++i) sp[i] = gp[i];
}

__device__ __forceinline__ void stage_bf16_T(const bf16_t* __restrict__ g, int ldg,
                                             bf16_t* __restrict__ sT) {
  const int t  = threadIdx.x;
  const int p  = t >> 2;
  const int cb = (t & 3) << 4;
  union { uint4 q[2]; unsigned short s[16]; } r0, r1;
  const uint4* g0 = (const uint4*)(g + (size_t)(2 * p) * ldg + cb);
  const uint4* g1 = (const uint4*)(g + (size_t)(2 * p + 1) * ldg + cb);
  r0.q[0] = g0[0]; r0.q[1] = g0[1];
  r1.q[0] = g1[0]; r1.q[1] = g1[1];
#pragma unroll
  for (int j = 0; j < 16; ++j) {
    u32 w = (u32)r0.s[j] | ((u32)r1.s[j] << 16);
    *(u32*)(sT + (size_t)(cb + j) * LDSS + 2 * p) = w;
  }
}

#ifdef HAVE_TDM
// TDM DMA of a 64x64 bf16 tile (row stride `stride` elems in memory) into LDS
// with pad_enable: +8 DWORDs per 32-DWORD row -> LDS row stride 80 elems.
__device__ __forceinline__ void tdm_load_tile(const bf16_t* gsrc, u32 lds_off,
                                              u32 stride) {
  const u64 ga = (u64)(uintptr_t)gsrc;
  u32x4 g0;
  g0[0] = 1u;                                  // count=1, user mode, no gather
  g0[1] = lds_off;                             // LDS byte address
  g0[2] = (u32)(ga & 0xffffffffu);             // global_addr[31:0]
  g0[3] = (u32)((ga >> 32) & 0x01ffffffu) | (2u << 30);  // addr[56:32] | type=2
  i32x8 g1;
  g1[0] = (int)((1u << 16)                     // data_size = 2 bytes
              | (1u << 20)                     // pad_enable
              | (4u << 22)                     // pad_interval: 32 DWORDs
              | (7u << 25));                   // pad_amount: 8 DWORDs
  g1[1] = (int)(64u << 16);                    // tensor_dim0 = 64 (lo16)
  g1[2] = (int)(64u << 16);                    // dim0 hi=0 | tensor_dim1=64 (lo16)
  g1[3] = (int)(64u << 16);                    // dim1 hi=0 | tile_dim0 = 64
  g1[4] = (int)64u;                            // tile_dim1 = 64, tile_dim2 = 0
  g1[5] = (int)stride;                         // tensor_dim0_stride[31:0]
  g1[6] = 0;
  g1[7] = 0;
  i32x4 z4 = {0, 0, 0, 0};
#if __clang_major__ >= 23
  i32x8 z8 = {0, 0, 0, 0, 0, 0, 0, 0};
  __builtin_amdgcn_tensor_load_to_lds(g0, g1, z4, z4, z8, 0);
#else
  __builtin_amdgcn_tensor_load_to_lds(g0, g1, z4, z4, 0);
#endif
}
#endif

// ---------------------------------------------------------------------------
// K0: RoPE table  tab[t][f] = {cos, sin}(t * 10000^(-2f/64)),  t<2048, f<32.
// ---------------------------------------------------------------------------
__global__ __launch_bounds__(128) void mha_rope_tab_k(float2* __restrict__ tab) {
  const int i = blockIdx.x * 128 + threadIdx.x;      // 0 .. 65535
  const int t = i >> 5, f = i & 31;
  const float inv = __expf(-(float)(2 * f) * (9.210340371976184f / 64.0f));
  float sn, cs;
  __sincosf((float)t * inv, &sn, &cs);
  tab[i] = make_float2(cs, sn);
}

// ---------------------------------------------------------------------------
// K1: QKV projection + RoPE.  grid=(64 Mtiles, 16 heads, 3 mats), block=128.
// ---------------------------------------------------------------------------
__global__ __launch_bounds__(128) void mha_qkv_rope_k(
    const float* __restrict__ x, const float* __restrict__ Wq,
    const float* __restrict__ Wk, const float* __restrict__ Wv,
    const float2* __restrict__ rope, bf16_t* __restrict__ qkv) {
  __shared__ __align__(16) bf16_t sA[64 * LDSS];    // x tile, row-major
  __shared__ __align__(16) bf16_t sBT[64 * LDSS];   // W tile, transposed

  const int mt = blockIdx.x;
  const int h  = blockIdx.y;
  const int z  = blockIdx.z;                        // 0=Q 1=K 2=V
  const float* W = (z == 0) ? Wq : (z == 1) ? Wk : Wv;

  const int wave = threadIdx.x >> 5;
  const int lane = threadIdx.x & 31;
  const int nl = lane & 15, kh = lane >> 4;
  const int m0 = wave * 16;

  v8f acc[4] = {};

  for (int kk = 0; kk < 1024; kk += 64) {
    __syncthreads();
    stage_f32_rm(x + (size_t)(mt * 64) * 1024 + kk, 1024, sA);
    stage_f32_T(W + (size_t)kk * 1024 + h * 64, 1024, sBT);
    __syncthreads();
    if (kk + 64 < 1024) {
      __builtin_prefetch(x + (size_t)(mt * 64 + (threadIdx.x >> 1)) * 1024 + kk + 64, 0, 1);
      __builtin_prefetch(W + (size_t)(kk + 64 + (threadIdx.x >> 1)) * 1024 + h * 64, 0, 1);
    }
#pragma unroll
    for (int kc = 0; kc < 2; ++kc) {
      v16bf a = load_frag(sA + m0 * LDSS + kc * 32, LDSS);
#pragma unroll
      for (int nt = 0; nt < 4; ++nt) {
        v16bf b = load_frag(sBT + (nt * 16) * LDSS + kc * 32, LDSS);
        acc[nt] = wmma_bf16(a, b, acc[nt]);
      }
    }
  }

  // Epilogue: table-driven RoPE on Q/K, 1/sqrt(D) on Q, paired b32 stores.
  const int rowbase = mt * 64 + m0 + kh * 8;
#pragma unroll
  for (int nt = 0; nt < 4; ++nt) {
    const int d = nt * 16 + nl;
#pragma unroll
    for (int r = 0; r < 8; ++r) {
      const int grow = rowbase + r;
      const int bb = grow >> 11;
      const int tt = grow & 2047;
      float val = acc[nt][r];
      float res = val;
      if (z < 2) {
        const float2 csn = rope[tt * 32 + (d >> 1)];
        const float part = __shfl_xor(val, 1, 32);
        res = (d & 1) ? (val * csn.x + part * csn.y)
                      : (val * csn.x - part * csn.y);
      }
      if (z == 0) res *= 0.125f;
      const float pres = __shfl_xor(res, 1, 32);
      if (!(d & 1)) {
        bf16_t* dst = qkv + (((size_t)(z * 2 + bb) * 16 + h) * 2048 + tt) * 64 + d;
        *(u32*)dst = pack2(res, pres);
      }
    }
  }
}

// ---------------------------------------------------------------------------
// K2: flash attention.  grid=(32 qblocks, 32 b*h), block=128 (4 waves x 16 q).
// K tiles: TDM into double-buffered LDS, prefetching block kb+1 while
// computing kb (TENSORcnt is in-order per wave -> s_wait_tensorcnt 1 means
// "current tile landed, prefetch may still fly").
// ---------------------------------------------------------------------------
__global__ __launch_bounds__(128) void mha_flash_attn_k(
    const bf16_t* __restrict__ qkv, bf16_t* __restrict__ attn) {
  __shared__ __align__(16) bf16_t sK[2][64 * LDSS];       // keys x d, row-major
  __shared__ __align__(16) bf16_t sVT[64 * LDSS];         // d x keys (transposed)
  __shared__ __align__(16) bf16_t sP[4 * 16 * LDSS];      // wave-private P

  const int qblk = blockIdx.x;
  const int bh   = blockIdx.y;
  const int b = bh >> 4, h = bh & 15;
  const size_t head_off = (((size_t)b * 16 + h) * 2048) * 64;
  const size_t mat = (size_t)2 * 16 * 2048 * 64;
  const bf16_t* Q  = qkv + head_off;
  const bf16_t* Kp = qkv + mat + head_off;
  const bf16_t* Vp = qkv + 2 * mat + head_off;

  const int wave = threadIdx.x >> 5;
  const int lane = threadIdx.x & 31;
  const int nl = lane & 15, kh = lane >> 4;
  const int q0 = qblk * 64 + wave * 16;

  v16bf aq0 = load_frag(Q + (size_t)q0 * 64, 64);
  v16bf aq1 = load_frag(Q + (size_t)q0 * 64 + 32, 64);

  v8f o[4] = {};
  float rm[8], rl[8];
#pragma unroll
  for (int r = 0; r < 8; ++r) { rm[r] = -3.0e38f; rl[r] = 0.f; }

  bf16_t* sPw = sP + wave * 16 * LDSS;
#ifdef HAVE_TDM
  const u32 sK_off0 = (u32)(uintptr_t)(void*)&sK[0][0];
  const u32 sK_off1 = (u32)(uintptr_t)(void*)&sK[1][0];
#endif

  for (int kb = 0; kb <= qblk; ++kb) {
    __syncthreads();                                   // prev tiles consumed
    stage_bf16_T(Vp + (size_t)kb * 64 * 64, 64, sVT);  // SW-transpose V
#ifdef HAVE_TDM
    if (wave == 0) {
      if (kb == 0)                                     // cold start: current
        tdm_load_tile(Kp, sK_off0, 64);
      if (kb < qblk)                                   // prefetch next K tile
        tdm_load_tile(Kp + (size_t)(kb + 1) * 64 * 64,
                      ((kb + 1) & 1) ? sK_off1 : sK_off0, 64);
      if (kb < qblk) __builtin_amdgcn_s_wait_tensorcnt(1);
      else           __builtin_amdgcn_s_wait_tensorcnt(0);
    }
#else
    stage_bf16_rm(Kp + (size_t)kb * 64 * 64, 64, sK[kb & 1]);
#endif
    __syncthreads();
    const bf16_t* sKc = sK[kb & 1];

    // S = Q K^T
    v8f sc[4];
#pragma unroll
    for (int nt = 0; nt < 4; ++nt) {
      v8f c = {};
      v16bf bk0 = load_frag(sKc + (nt * 16) * LDSS, LDSS);
      v16bf bk1 = load_frag(sKc + (nt * 16) * LDSS + 32, LDSS);
      c = wmma_bf16(aq0, bk0, c);
      c = wmma_bf16(aq1, bk1, c);
      sc[nt] = c;
    }

    if (kb == qblk) {                         // causal mask on diagonal block
#pragma unroll
      for (int nt = 0; nt < 4; ++nt)
#pragma unroll
        for (int r = 0; r < 8; ++r) {
          const int key = nt * 16 + nl;
          const int qr  = wave * 16 + kh * 8 + r;
          if (key > qr) sc[nt][r] = -3.0e38f;
        }
    }

    // Online softmax (rows across 16 lanes of a half-wave).
#pragma unroll
    for (int r = 0; r < 8; ++r) {
      float m = fmaxf(fmaxf(sc[0][r], sc[1][r]), fmaxf(sc[2][r], sc[3][r]));
#pragma unroll
      for (int off = 1; off < 16; off <<= 1)
        m = fmaxf(m, __shfl_xor(m, off, 32));
      const float mnew = fmaxf(rm[r], m);
      const float corr = __expf(rm[r] - mnew);
      rm[r] = mnew;
      float rs = 0.f;
#pragma unroll
      for (int nt = 0; nt < 4; ++nt) {
        const float e = __expf(sc[nt][r] - mnew);
        sc[nt][r] = e;
        rs += e;
      }
#pragma unroll
      for (int off = 1; off < 16; off <<= 1)
        rs += __shfl_xor(rs, off, 32);
      rl[r] = rl[r] * corr + rs;
#pragma unroll
      for (int dt = 0; dt < 4; ++dt) o[dt][r] *= corr;
    }

    // P -> wave-private LDS as packed 32-bit pairs (DS in-order per wave).
#pragma unroll
    for (int nt = 0; nt < 4; ++nt)
#pragma unroll
      for (int r = 0; r < 8; ++r) {
        const float v = sc[nt][r];
        const float pr = __shfl_xor(v, 1, 32);
        if (!(nl & 1))
          *(u32*)(sPw + (kh * 8 + r) * LDSS + nt * 16 + nl) = pack2(v, pr);
      }

    v16bf ap0 = load_frag(sPw, LDSS);
    v16bf ap1 = load_frag(sPw + 32, LDSS);
#pragma unroll
    for (int dt = 0; dt < 4; ++dt) {
      v16bf bv0 = load_frag(sVT + (dt * 16) * LDSS, LDSS);
      v16bf bv1 = load_frag(sVT + (dt * 16) * LDSS + 32, LDSS);
      o[dt] = wmma_bf16(ap0, bv0, o[dt]);
      o[dt] = wmma_bf16(ap1, bv1, o[dt]);
    }
  }

  // Normalize and store packed pairs into (B*T, C).
#pragma unroll
  for (int r = 0; r < 8; ++r) {
    const float rinv = 1.0f / rl[r];
    const int qg = q0 + kh * 8 + r;
    bf16_t* dst = attn + ((size_t)b * 2048 + qg) * 1024 + h * 64;
#pragma unroll
    for (int dt = 0; dt < 4; ++dt) {
      const float v = o[dt][r] * rinv;
      const float pr = __shfl_xor(v, 1, 32);
      if (!(nl & 1)) *(u32*)(dst + dt * 16 + nl) = pack2(v, pr);
    }
  }
}

// ---------------------------------------------------------------------------
// K3: output projection + bias.  grid=(64, 16), block=128.  A tile via TDM.
// ---------------------------------------------------------------------------
__global__ __launch_bounds__(128) void mha_out_proj_k(
    const bf16_t* __restrict__ a, const float* __restrict__ Wo,
    const float* __restrict__ bo, float* __restrict__ out) {
  __shared__ __align__(16) bf16_t sA[64 * LDSS];
  __shared__ __align__(16) bf16_t sBT[64 * LDSS];

  const int mt = blockIdx.x;
  const int ntile = blockIdx.y;
  const int wave = threadIdx.x >> 5;
  const int lane = threadIdx.x & 31;
  const int nl = lane & 15, kh = lane >> 4;
  const int m0 = wave * 16;
#ifdef HAVE_TDM
  const u32 sA_off = (u32)(uintptr_t)(void*)sA;
#endif

  v8f acc[4] = {};

  for (int kk = 0; kk < 1024; kk += 64) {
    __syncthreads();
#ifdef HAVE_TDM
    if (wave == 0) tdm_load_tile(a + (size_t)(mt * 64) * 1024 + kk, sA_off, 1024);
#else
    stage_bf16_rm(a + (size_t)(mt * 64) * 1024 + kk, 1024, sA);
#endif
    stage_f32_T(Wo + (size_t)kk * 1024 + ntile * 64, 1024, sBT);
#ifdef HAVE_TDM
    if (wave == 0) __builtin_amdgcn_s_wait_tensorcnt(0);
#endif
    __syncthreads();
    if (kk + 64 < 1024)
      __builtin_prefetch(Wo + (size_t)(kk + 64 + (threadIdx.x >> 1)) * 1024 + ntile * 64, 0, 1);
#pragma unroll
    for (int kc = 0; kc < 2; ++kc) {
      v16bf af = load_frag(sA + m0 * LDSS + kc * 32, LDSS);
#pragma unroll
      for (int nt = 0; nt < 4; ++nt) {
        v16bf bf = load_frag(sBT + (nt * 16) * LDSS + kc * 32, LDSS);
        acc[nt] = wmma_bf16(af, bf, acc[nt]);
      }
    }
  }

  const int rowbase = mt * 64 + m0 + kh * 8;
#pragma unroll
  for (int nt = 0; nt < 4; ++nt) {
    const int col = ntile * 64 + nt * 16 + nl;
    const float bias = bo[col];
#pragma unroll
    for (int r = 0; r < 8; ++r) {
      const float v = acc[nt][r] + bias;
      const float pr = __shfl_xor(v, 1, 32);
      if (!(nl & 1)) {
        float2 w = make_float2(v, pr);
        *(float2*)(out + (size_t)(rowbase + r) * 1024 + col) = w;
      }
    }
  }
}

// ---------------------------------------------------------------------------
extern "C" void kernel_launch(void* const* d_in, const int* in_sizes, int n_in,
                              void* d_out, int out_size, void* d_ws, size_t ws_size,
                              hipStream_t stream) {
  (void)in_sizes; (void)n_in; (void)out_size; (void)ws_size;
  const float* x  = (const float*)d_in[0];
  const float* Wq = (const float*)d_in[1];
  const float* Wk = (const float*)d_in[2];
  const float* Wv = (const float*)d_in[3];
  const float* Wo = (const float*)d_in[4];
  const float* bo = (const float*)d_in[5];
  float* out = (float*)d_out;

  bf16_t* qkv  = (bf16_t*)d_ws;                          // 24 MB
  bf16_t* attn = qkv + (size_t)3 * 2 * 16 * 2048 * 64;   // 8 MB
  float2* rope = (float2*)(attn + (size_t)2 * 2048 * 1024);  // 512 KB

  mha_rope_tab_k<<<512, 128, 0, stream>>>(rope);
  mha_qkv_rope_k<<<dim3(64, 16, 3), 128, 0, stream>>>(x, Wq, Wk, Wv, rope, qkv);
  mha_flash_attn_k<<<dim3(32, 32), 128, 0, stream>>>(qkv, attn);
  mha_out_proj_k<<<dim3(64, 16), 128, 0, stream>>>(attn, Wo, bo, out);
}
